// TemporalAttentionModule_4750233830133
// MI455X (gfx1250) — matrix operand order repaired and checked by reading, not compiled, not run on hardware
//
#include <hip/hip_runtime.h>
#include <hip/hip_bf16.h>
#include <math.h>

// ---------------- problem constants ----------------
#define DIMN   1024
#define HEADS  16
#define HSZ    64
#define BATCH  2
#define TLEN   4096
#define MROWS  (BATCH * TLEN)   // 8192 rows

typedef __attribute__((ext_vector_type(16))) __bf16 v16bf_t;
typedef __attribute__((ext_vector_type(8)))  float  v8f_t;

// round-to-nearest-even fp32 -> bf16 (pure integer; avoids host-pass __bf16 math)
__device__ inline unsigned short f2bf(float f) {
  unsigned int u = __builtin_bit_cast(unsigned int, f);
  unsigned int lsb = (u >> 16) & 1u;
  u += 0x7fffu + lsb;
  return (unsigned short)(u >> 16);
}

// ---- WMMA fragment loads (bf16, wave32) ----
// A-matrix 16x32: lane L (r=L%16,h=L/16) row M=r holds K = {8h..8h+7} then {16+8h..16+8h+7}
__device__ inline v16bf_t load_frag_a(const unsigned short* p, int h) {
  union { v16bf_t v; uint4 q[2]; } f;
  const unsigned short* s = p + 8 * h;
  f.q[0] = *(const uint4*)(s);
  f.q[1] = *(const uint4*)(s + 16);
  return f.v;
}
// B-matrix 32x16 sourced from Wt[N,K] row-major: lane L (n=L%16,h=L/16) holds K = {8h..8h+7} then {16+8h..16+8h+7}
__device__ inline v16bf_t load_frag_b(const unsigned short* p, int h) {
  union { v16bf_t v; uint4 q[2]; } f;
  const unsigned short* s = p + 8 * h;
  f.q[0] = *(const uint4*)(s);
  f.q[1] = *(const uint4*)(s + 16);
  return f.v;
}

#define WMMA_BF16(a, b, c) \
  __builtin_amdgcn_wmma_f32_16x16x32_bf16(false, (a), false, (b), (short)0, (c), false, false)

// ---------------- WMMA GEMM: C[M,N] = A[M,K](bf16) * Wt[N,K](bf16)^T + bias ----------------
// Each wave computes a 32x32 output tile (2x2 register tiling, 4 accumulators).
// Block = 8 waves (2 x 4) -> block tile 64(M) x 128(N). K-loop is software-pipelined:
// next K-step fragments are issued before the current WMMAs so loads overlap compute.
// EPI: 0 = bias only, 1 = bias + exact GELU
template <int EPI>
__global__ __launch_bounds__(256) void gemm_bf16_wmma(
    const unsigned short* __restrict__ A, int lda,
    const unsigned short* __restrict__ Bt, int ldb,
    const float* __restrict__ bias,
    float* __restrict__ C, int ldc, int K) {
  const int wave = threadIdx.x >> 5;
  const int lane = threadIdx.x & 31;
  const int h = lane >> 4;
  const int r = lane & 15;

  const int mBase = blockIdx.y * 64 + (wave >> 2) * 32;
  const int nBase = blockIdx.x * 128 + (wave & 3) * 32;

  const unsigned short* aRow0 = A + (size_t)(mBase + r) * lda;
  const unsigned short* aRow1 = A + (size_t)(mBase + 16 + r) * lda;
  const unsigned short* bRow0 = Bt + (size_t)(nBase + r) * ldb;
  const unsigned short* bRow1 = Bt + (size_t)(nBase + 16 + r) * ldb;

  v8f_t acc00 = {}, acc01 = {}, acc10 = {}, acc11 = {};

  // prologue: first fragments
  v16bf_t a0 = load_frag_a(aRow0, h);
  v16bf_t a1 = load_frag_a(aRow1, h);
  v16bf_t b0 = load_frag_b(bRow0, h);
  v16bf_t b1 = load_frag_b(bRow1, h);

  for (int k0 = 32; k0 < K; k0 += 32) {
    // issue next-iteration loads before consuming current fragments
    v16bf_t na0 = load_frag_a(aRow0 + k0, h);
    v16bf_t na1 = load_frag_a(aRow1 + k0, h);
    v16bf_t nb0 = load_frag_b(bRow0 + k0, h);
    v16bf_t nb1 = load_frag_b(bRow1 + k0, h);
    if (k0 + 32 < K) {
      __builtin_prefetch((const void*)(aRow0 + k0 + 32), 0, 1);
      __builtin_prefetch((const void*)(bRow0 + k0 + 32), 0, 1);
    }
    acc00 = WMMA_BF16(a0, b0, acc00);
    acc01 = WMMA_BF16(a0, b1, acc01);
    acc10 = WMMA_BF16(a1, b0, acc10);
    acc11 = WMMA_BF16(a1, b1, acc11);
    a0 = na0; a1 = na1; b0 = nb0; b1 = nb1;
  }
  // epilogue K-step
  acc00 = WMMA_BF16(a0, b0, acc00);
  acc01 = WMMA_BF16(a0, b1, acc01);
  acc10 = WMMA_BF16(a1, b0, acc10);
  acc11 = WMMA_BF16(a1, b1, acc11);

  // C/D layout: lane holds column n = r (within 16-tile); rows m = 8h + j
  const v8f_t* accs[4] = {&acc00, &acc01, &acc10, &acc11};
#pragma unroll
  for (int mt = 0; mt < 2; ++mt) {
#pragma unroll
    for (int nt = 0; nt < 2; ++nt) {
      const v8f_t& acc = *accs[mt * 2 + nt];
      const int gn = nBase + nt * 16 + r;
      const float bb = bias[gn];
      float* cCol = C + gn;
#pragma unroll
      for (int j = 0; j < 8; ++j) {
        const int gm = mBase + mt * 16 + 8 * h + j;
        float x = acc[j] + bb;
        if (EPI == 1) x = 0.5f * x * (1.0f + erff(x * 0.70710678118654752f));
        cCol[(size_t)gm * ldc] = x;
      }
    }
  }
}

// ---------------- LayerNorm (optional residual), fp32 + bf16 outputs ----------------
__global__ __launch_bounds__(256) void layernorm_kernel(
    const float* __restrict__ X, const float* __restrict__ R,
    const float* __restrict__ g, const float* __restrict__ b,
    float* __restrict__ outF, unsigned short* __restrict__ outH) {
  __shared__ float sSum[256], sSq[256];
  const int row = blockIdx.x;
  const int tid = threadIdx.x;
  const float* x = X + (size_t)row * DIMN;
  const float* r = R ? (R + (size_t)row * DIMN) : nullptr;

  float v[4];
  float s = 0.f, sq = 0.f;
#pragma unroll
  for (int i = 0; i < 4; ++i) {
    const int c = tid + i * 256;
    float t = x[c];
    if (r) t += r[c];
    v[i] = t;
    s += t;
    sq += t * t;
  }
  sSum[tid] = s;
  sSq[tid] = sq;
  __syncthreads();
  for (int st = 128; st > 0; st >>= 1) {
    if (tid < st) { sSum[tid] += sSum[tid + st]; sSq[tid] += sSq[tid + st]; }
    __syncthreads();
  }
  const float mean = sSum[0] * (1.0f / DIMN);
  const float var  = sSq[0] * (1.0f / DIMN) - mean * mean;
  const float inv  = rsqrtf(var + 1e-6f);
#pragma unroll
  for (int i = 0; i < 4; ++i) {
    const int c = tid + i * 256;
    const float y = (v[i] - mean) * inv * g[c] + b[c];
    if (outF) outF[(size_t)row * DIMN + c] = y;
    outH[(size_t)row * DIMN + c] = f2bf(y);
  }
}

// ---------------- weight transpose + fp32->bf16: Wt[N,K] = W[K,N] ----------------
__global__ __launch_bounds__(256) void conv_transpose_kernel(
    const float* __restrict__ W, unsigned short* __restrict__ Wt, int K, int N) {
  const int idx = blockIdx.x * 256 + threadIdx.x;
  if (idx >= K * N) return;
  const int k = idx / N;
  const int n = idx % N;
  Wt[(size_t)n * K + k] = f2bf(W[idx]);
}

// ---------------- windowed local attention (S=3, query = window index 1) ----------------
__device__ inline float waveSum(float v) {
#pragma unroll
  for (int m = 16; m > 0; m >>= 1) v += __shfl_xor(v, m, 32);
  return v;
}

__device__ inline float2 ld2tv(const float* base, int b, int tt, int off) {
  if (tt < 0 || tt >= TLEN) return make_float2(0.f, 0.f);
  const float* p = base + ((size_t)b * TLEN + tt) * DIMN + off;
  return make_float2(p[0], p[1]);
}

__global__ __launch_bounds__(256) void local_attn_kernel(
    const float* __restrict__ Q, const float* __restrict__ K,
    const float* __restrict__ V, unsigned short* __restrict__ OutCat) {
  const int wave = threadIdx.x >> 5;
  const int lane = threadIdx.x & 31;
  int idx = blockIdx.x * 8 + wave;       // (b*T + t)*HEADS + h
  const int h = idx % HEADS;  idx /= HEADS;
  const int t = idx % TLEN;
  const int b = idx / TLEN;
  const int off = h * HSZ + lane * 2;

  const size_t obase = ((size_t)(b * TLEN + t)) * (2 * DIMN) + h * HSZ + lane * 2;

#pragma unroll
  for (int dir = 0; dir < 2; ++dir) {
    const int sgn = dir ? 1 : -1;        // fwd: t - j, rev: t + j
    const float2 qv = ld2tv(Q, b, t + sgn, off);  // query = window index 1
    float sc[3];
#pragma unroll
    for (int j = 0; j < 3; ++j) {
      const float2 kv = ld2tv(K, b, t + sgn * j, off);
      sc[j] = waveSum(qv.x * kv.x + qv.y * kv.y) * 0.125f;  // 1/sqrt(64)
    }
    const float mx = fmaxf(sc[0], fmaxf(sc[1], sc[2]));
    const float e0 = __expf(sc[0] - mx);
    const float e1 = __expf(sc[1] - mx);
    const float e2 = __expf(sc[2] - mx);
    const float inv = 1.0f / (e0 + e1 + e2);
    float ox = 0.f, oy = 0.f;
    const float p_[3] = {e0 * inv, e1 * inv, e2 * inv};
#pragma unroll
    for (int j = 0; j < 3; ++j) {
      const float2 vv = ld2tv(V, b, t + sgn * j, off);
      ox += p_[j] * vv.x;
      oy += p_[j] * vv.y;
    }
    const size_t o = obase + (size_t)dir * DIMN;
    OutCat[o]     = f2bf(ox);
    OutCat[o + 1] = f2bf(oy);
  }
}

// ---------------- host launcher ----------------
extern "C" void kernel_launch(void* const* d_in, const int* in_sizes, int n_in,
                              void* d_out, int out_size, void* d_ws, size_t ws_size,
                              hipStream_t stream) {
  const float* x   = (const float*)d_in[0];
  const float* Wq  = (const float*)d_in[1];
  const float* bq  = (const float*)d_in[2];
  const float* Wk  = (const float*)d_in[3];
  const float* bk  = (const float*)d_in[4];
  const float* Wv  = (const float*)d_in[5];
  const float* bv  = (const float*)d_in[6];
  const float* Wfb = (const float*)d_in[7];
  const float* bfb = (const float*)d_in[8];
  const float* Wo  = (const float*)d_in[9];
  const float* bo  = (const float*)d_in[10];
  const float* g1  = (const float*)d_in[11];
  const float* b1  = (const float*)d_in[12];
  const float* g2  = (const float*)d_in[13];
  const float* b2  = (const float*)d_in[14];
  float* out = (float*)d_out;

  char* wsp = (char*)d_ws;
  auto take = [&](size_t bytes) -> void* {
    void* p = (void*)wsp;
    wsp += (bytes + 255) & ~(size_t)255;
    return p;
  };

  const size_t MD_F = (size_t)MROWS * DIMN * sizeof(float);
  const size_t MD_H = (size_t)MROWS * DIMN * sizeof(unsigned short);

  float*          xnF  = (float*)take(MD_F);            // layernorm1 out (residual)
  unsigned short* xnH  = (unsigned short*)take(MD_H);   // bf16 copy for QKV gemms
  unsigned short* WqT  = (unsigned short*)take((size_t)DIMN * DIMN * 2);
  unsigned short* WkT  = (unsigned short*)take((size_t)DIMN * DIMN * 2);
  unsigned short* WvT  = (unsigned short*)take((size_t)DIMN * DIMN * 2);
  unsigned short* WoT  = (unsigned short*)take((size_t)DIMN * DIMN * 2);
  unsigned short* WfbT = (unsigned short*)take((size_t)DIMN * 2 * DIMN * 2);
  float*          qB   = (float*)take(MD_F);
  float*          kB   = (float*)take(MD_F);
  float*          vB   = (float*)take(MD_F);
  unsigned short* cat  = (unsigned short*)take((size_t)MROWS * 2 * DIMN * 2);
  // aliases (lifetimes disjoint): y reuses qB after attention, zH reuses xnH after QKV gemms
  float*          yB = qB;
  unsigned short* zH = xnH;

  const dim3 blk(256);
  const dim3 gemmGrid(DIMN / 128, MROWS / 64);   // (8, 128), 64x128 block tile

  // 1) layernorm1 -> xnF (fp32 residual), xnH (bf16)
  layernorm_kernel<<<MROWS, blk, 0, stream>>>(x, nullptr, g1, b1, xnF, xnH);

  // 2) weight transpose/convert
  conv_transpose_kernel<<<(DIMN * DIMN + 255) / 256, blk, 0, stream>>>(Wq, WqT, DIMN, DIMN);
  conv_transpose_kernel<<<(DIMN * DIMN + 255) / 256, blk, 0, stream>>>(Wk, WkT, DIMN, DIMN);
  conv_transpose_kernel<<<(DIMN * DIMN + 255) / 256, blk, 0, stream>>>(Wv, WvT, DIMN, DIMN);
  conv_transpose_kernel<<<(DIMN * DIMN + 255) / 256, blk, 0, stream>>>(Wo, WoT, DIMN, DIMN);
  conv_transpose_kernel<<<(2 * DIMN * DIMN + 255) / 256, blk, 0, stream>>>(Wfb, WfbT, 2 * DIMN, DIMN);

  // 3) QKV projections (WMMA bf16 -> fp32)
  gemm_bf16_wmma<0><<<gemmGrid, blk, 0, stream>>>(xnH, DIMN, WqT, DIMN, bq, qB, DIMN, DIMN);
  gemm_bf16_wmma<0><<<gemmGrid, blk, 0, stream>>>(xnH, DIMN, WkT, DIMN, bk, kB, DIMN, DIMN);
  gemm_bf16_wmma<0><<<gemmGrid, blk, 0, stream>>>(xnH, DIMN, WvT, DIMN, bv, vB, DIMN, DIMN);

  // 4) windowed local attention -> cat[M, 2*DIM] bf16 (fwd | rev)
  local_attn_kernel<<<(MROWS * HEADS) / 8, blk, 0, stream>>>(qB, kB, vB, cat);

  // 5) feed-back projection: cat @ Wfb + bfb -> yB (fp32)
  gemm_bf16_wmma<0><<<gemmGrid, blk, 0, stream>>>(cat, 2 * DIMN, WfbT, 2 * DIMN, bfb, yB, DIMN, 2 * DIMN);

  // 6) layernorm2(y + xn) -> zH (bf16 only)
  layernorm_kernel<<<MROWS, blk, 0, stream>>>(yB, xnF, g2, b2, nullptr, zH);

  // 7) output projection + exact GELU -> d_out (fp32)
  gemm_bf16_wmma<1><<<gemmGrid, blk, 0, stream>>>(zH, DIMN, WoT, DIMN, bo, out, DIMN, DIMN);

  (void)in_sizes; (void)n_in; (void)out_size; (void)ws_size;
}
